// UniCoverageLSTM_90340342104609
// MI455X (gfx1250) — matrix operand order repaired and checked by reading, not compiled, not run on hardware
//
#include <hip/hip_runtime.h>

// ---------------------------------------------------------------------------
// CDNA5 (gfx1250) wave32 WMMA implementation of the coverage-attention LSTM.
// All GEMMs use v_wmma_f32_16x16x32_bf16 (fp32 operands downconverted to bf16
// while staging through LDS; fp32 accumulate). Big GEMM is double-buffered.
// ---------------------------------------------------------------------------

typedef __attribute__((ext_vector_type(16))) __bf16 v16bf;
typedef __attribute__((ext_vector_type(2)))  __bf16 v2bf;
typedef __attribute__((ext_vector_type(8)))  float  v8f;

#define T_STEPS  64
#define BATCH    32
#define S_LEN    512
#define NEMB_    512
#define NHID_    512
#define SRC_HID_ 1024
#define ATTN_    512
#define NG_      2048   // 4*NHID

__device__ __forceinline__ float sigmoidf_(float x) {
  return 1.0f / (1.0f + __expf(-x));
}

__device__ __forceinline__ float fast_tanhf(float x) {
#if __has_builtin(__builtin_amdgcn_tanhf)
  return __builtin_amdgcn_tanhf(x);        // v_tanh_f32 (gfx1250)
#else
  float e = __expf(2.0f * x);              // v_exp_f32 based fallback
  return 1.0f - 2.0f / (e + 1.0f);
#endif
}

// packed bf16 pair store into LDS (4B aligned) -> v_cvt_pk_bf16_f32 + ds_store_b32
__device__ __forceinline__ void st_bf16x2(__bf16* dst, float f0, float f1) {
  v2bf v;
  v[0] = (__bf16)f0;
  v[1] = (__bf16)f1;
  *reinterpret_cast<v2bf*>(dst) = v;
}

// --- WMMA fragment loaders (ISA 7.12.2 layouts, wave32) --------------------
// A fragment: 16x32 bf16 tile, row-major in LDS with leading dim ld (16B-aligned rows).
// lane<16: row=lane, K = {0..7, 16..23}; lane>=16: row=lane-16, K={8..15,24..31}
__device__ __forceinline__ v16bf frag_a(const __bf16* p, int ld, int lane) {
  const int r  = lane & 15;
  const int h8 = (lane >> 4) << 3;      // 0 or 8
  const __bf16* row = p + r * ld;
  v16bf a;
#pragma unroll
  for (int j = 0; j < 8; ++j) a[j]     = row[h8 + j];
#pragma unroll
  for (int j = 0; j < 8; ++j) a[8 + j] = row[16 + h8 + j];
  return a;
}

// B fragment: 32x16 bf16 tile stored TRANSPOSED in LDS (bt[n][k], leading ld).
// lane<16: col=lane, K = {0..7, 16..23}; lane>=16: col=lane-16, K={8..15,24..31}
__device__ __forceinline__ v16bf frag_b(const __bf16* p, int ld, int lane) {
  const int n   = lane & 15;
  const int h8  = (lane >> 4) << 3;     // 0 or 8
  const __bf16* col = p + n * ld;
  v16bf b;
#pragma unroll
  for (int j = 0; j < 8; ++j) b[j]     = col[h8 + j];
#pragma unroll
  for (int j = 0; j < 8; ++j) b[8 + j] = col[16 + h8 + j];
  return b;
}

// ---------------------------------------------------------------------------
// e_h = (enc @ Wh + Wh_b) * source_mask     M=B*S=16384, K=1024, N=512
// block tile 64x64, 8 waves (4x2), two 16x16 accumulators per wave.
// Double-buffered LDS; register prefetch of the next K-tile; 1 barrier/iter.
// ---------------------------------------------------------------------------
__global__ __launch_bounds__(256)
void k_gemm_eh(const float* __restrict__ enc, const float* __restrict__ Wh,
               const float* __restrict__ Whb, const float* __restrict__ smask,
               float* __restrict__ eh) {
  __shared__ __bf16 lds_a[2][64 * 40];
  __shared__ __bf16 lds_bt[2][64 * 40];
  const int tid  = threadIdx.x;
  const int lane = tid & 31, wid = tid >> 5;
  const int m0 = blockIdx.y * 64;
  const int n0 = blockIdx.x * 64;
  const int wr = (wid & 3) * 16;   // wave row within tile
  const int wc = (wid >> 2) * 32;  // wave col within tile
  v8f acc0 = {0.f,0.f,0.f,0.f,0.f,0.f,0.f,0.f};
  v8f acc1 = {0.f,0.f,0.f,0.f,0.f,0.f,0.f,0.f};

  float2 ar[4], br[4];
  // prefetch K-tile 0 into registers
#pragma unroll
  for (int j = 0; j < 4; ++j) {
    int p = tid + 256 * j;
    { int r = p >> 4, c = (p & 15) << 1;                       // A: 64x32
      ar[j] = *reinterpret_cast<const float2*>(
          &enc[(size_t)(m0 + r) * SRC_HID_ + c]); }
    { int n = p & 63, kr = (p >> 6) << 1;                      // B: 32x64
      br[j].x = Wh[(size_t)kr       * ATTN_ + n0 + n];
      br[j].y = Wh[(size_t)(kr + 1) * ATTN_ + n0 + n]; }
  }

  int buf = 0;
  for (int k0 = 0; k0 < SRC_HID_; k0 += 32) {
    // commit staged registers to LDS (packed bf16 pairs)
#pragma unroll
    for (int j = 0; j < 4; ++j) {
      int p = tid + 256 * j;
      { int r = p >> 4, c = (p & 15) << 1;
        st_bf16x2(&lds_a[buf][r * 40 + c], ar[j].x, ar[j].y); }
      { int n = p & 63, kr = (p >> 6) << 1;
        st_bf16x2(&lds_bt[buf][n * 40 + kr], br[j].x, br[j].y); }
    }
    __syncthreads();
    // prefetch next K-tile while this one computes
    if (k0 + 32 < SRC_HID_) {
      const int kn = k0 + 32;
#pragma unroll
      for (int j = 0; j < 4; ++j) {
        int p = tid + 256 * j;
        { int r = p >> 4, c = (p & 15) << 1;
          ar[j] = *reinterpret_cast<const float2*>(
              &enc[(size_t)(m0 + r) * SRC_HID_ + kn + c]); }
        { int n = p & 63, kr = (p >> 6) << 1;
          br[j].x = Wh[(size_t)(kn + kr)     * ATTN_ + n0 + n];
          br[j].y = Wh[(size_t)(kn + kr + 1) * ATTN_ + n0 + n]; }
      }
    }
    v16bf a  = frag_a(&lds_a[buf][wr * 40], 40, lane);
    v16bf b0 = frag_b(&lds_bt[buf][wc * 40], 40, lane);
    v16bf b1 = frag_b(&lds_bt[buf][(wc + 16) * 40], 40, lane);
    acc0 = __builtin_amdgcn_wmma_f32_16x16x32_bf16(false, a, false, b0, (short)0, acc0, false, false);
    acc1 = __builtin_amdgcn_wmma_f32_16x16x32_bf16(false, a, false, b1, (short)0, acc1, false, false);
    buf ^= 1;
  }
  const int col = lane & 15;
  const int rb  = (lane >> 4) << 3;
#pragma unroll
  for (int r = 0; r < 8; ++r) {
    int gm = m0 + wr + rb + r;               // gm == b*S + s
    float msk = smask[gm];
    int gn0 = n0 + wc + col;
    eh[(size_t)gm * ATTN_ + gn0]      = (acc0[r] + Whb[gn0]) * msk;
    eh[(size_t)gm * ATTN_ + gn0 + 16] = (acc1[r] + Whb[gn0 + 16]) * msk;
  }
}

// ---------------------------------------------------------------------------
// e_s = (h @ Ws) * mask[:,t]      M=32, K=512, N=512. block tile 32x64.
// ---------------------------------------------------------------------------
__global__ __launch_bounds__(256)
void k_es(const float* __restrict__ h, const float* __restrict__ Ws,
          const float* __restrict__ mask, int t, float* __restrict__ es) {
  __shared__ __bf16 lds_a[32 * 40];
  __shared__ __bf16 lds_bt[64 * 40];
  const int tid  = threadIdx.x;
  const int lane = tid & 31, wid = tid >> 5;
  const int n0 = blockIdx.x * 64;
  const int wr = (wid & 1) * 16;
  const int wc = (wid >> 1) * 16;
  v8f acc = {0.f,0.f,0.f,0.f,0.f,0.f,0.f,0.f};

  for (int k0 = 0; k0 < NHID_; k0 += 32) {
#pragma unroll
    for (int j = 0; j < 2; ++j) {            // A: 32x32 (512 pairs)
      int p = tid + 256 * j;
      int r = p >> 4, c = (p & 15) << 1;
      const float2 v = *reinterpret_cast<const float2*>(&h[r * NHID_ + k0 + c]);
      st_bf16x2(&lds_a[r * 40 + c], v.x, v.y);
    }
#pragma unroll
    for (int j = 0; j < 4; ++j) {            // B: 32x64 transposed (1024 pairs)
      int p = tid + 256 * j;
      int n = p & 63, kr = (p >> 6) << 1;
      float b0 = Ws[(size_t)(k0 + kr)     * ATTN_ + n0 + n];
      float b1 = Ws[(size_t)(k0 + kr + 1) * ATTN_ + n0 + n];
      st_bf16x2(&lds_bt[n * 40 + kr], b0, b1);
    }
    __syncthreads();
    v16bf a = frag_a(&lds_a[wr * 40], 40, lane);
    v16bf b = frag_b(&lds_bt[wc * 40], 40, lane);
    acc = __builtin_amdgcn_wmma_f32_16x16x32_bf16(false, a, false, b, (short)0, acc, false, false);
    __syncthreads();
  }
  const int col = lane & 15;
  const int rb  = (lane >> 4) << 3;
#pragma unroll
  for (int r = 0; r < 8; ++r) {
    int b_ = wr + rb + r;                    // batch index
    float m = mask[b_ * T_STEPS + t];
    es[b_ * ATTN_ + n0 + wc + col] = acc[r] * m;
  }
}

// ---------------------------------------------------------------------------
// logit[b,s] = (v . tanh(e_h + e_s + cov*Wc*smask)) * smask
// one wave per (b,s) row; wave32 shuffle reduction; v_tanh_f32 fast path
// ---------------------------------------------------------------------------
__global__ __launch_bounds__(256)
void k_attn(const float* __restrict__ eh, const float* __restrict__ es,
            const float* __restrict__ cov, const float* __restrict__ Wc,
            const float* __restrict__ vw, const float* __restrict__ smask,
            int t, float* __restrict__ logit, float* __restrict__ logit_out) {
  const int lane = threadIdx.x & 31;
  const int wid  = threadIdx.x >> 5;
  const int row  = blockIdx.x * 8 + wid;     // row = b*S + s
  const int b = row >> 9, s = row & 511;
  const float cv = cov[row];
  const float sm = smask[row];
  const float* ehr = eh + (size_t)row * ATTN_;
  const float* esr = es + b * ATTN_;
  float acc = 0.f;
#pragma unroll
  for (int j = 0; j < 16; ++j) {
    int a = lane + 32 * j;
    float e = fast_tanhf(ehr[a] + esr[a] + cv * Wc[a] * sm);
    acc += e * vw[a];
  }
#pragma unroll
  for (int off = 16; off > 0; off >>= 1) acc += __shfl_xor(acc, off, 32);
  if (lane == 0) {
    float lg = acc * sm;
    logit[row] = lg;
    logit_out[(size_t)b * T_STEPS * S_LEN + (size_t)t * S_LEN + s] = lg;
  }
}

// ---------------------------------------------------------------------------
// masked softmax over S, coverage update, ctx = alpha @ enc. block per batch.
// ---------------------------------------------------------------------------
__global__ __launch_bounds__(512)
void k_soft_ctx(const float* __restrict__ logit, const float* __restrict__ smask,
                const float* __restrict__ enc, int t,
                float* __restrict__ cov, float* __restrict__ attn_out,
                float* __restrict__ ctx) {
  __shared__ float red[512];
  __shared__ float alpha_s[512];
  const int b = blockIdx.x;
  const int s = threadIdx.x;
  const float m = smask[b * S_LEN + s];
  float x = logit[b * S_LEN + s];
  x = fminf(fmaxf(x, -15.f), 15.f) * m;
  red[s] = x;
  __syncthreads();
  for (int off = 256; off > 0; off >>= 1) {
    if (s < off) red[s] = fmaxf(red[s], red[s + off]);
    __syncthreads();
  }
  const float mx = red[0];
  __syncthreads();
  float e = __expf(x - mx) * m;
  red[s] = e;
  __syncthreads();
  for (int off = 256; off > 0; off >>= 1) {
    if (s < off) red[s] += red[s + off];
    __syncthreads();
  }
  const float al = e / (red[0] + 1e-6f);
  alpha_s[s] = al;
  attn_out[(size_t)b * T_STEPS * S_LEN + (size_t)t * S_LEN + s] = al;
  cov[b * S_LEN + s] += al;
  __syncthreads();
  // ctx: each thread accumulates dims s and s+512 over all source positions
  float a0 = 0.f, a1 = 0.f;
  const float* eb = enc + (size_t)b * S_LEN * SRC_HID_;
  for (int ss = 0; ss < S_LEN; ++ss) {
    const float av = alpha_s[ss];
    a0 += av * eb[(size_t)ss * SRC_HID_ + s];
    a1 += av * eb[(size_t)ss * SRC_HID_ + s + 512];
  }
  ctx[b * SRC_HID_ + s]       = a0;
  ctx[b * SRC_HID_ + s + 512] = a1;
}

// ---------------------------------------------------------------------------
// pre = x_t @ W_ih + h @ W_hh + ctx @ W_ch + bias  (segmented-K WMMA GEMM)
// M=32, Kcat=2048, N=2048. block tile 32x64.
// ---------------------------------------------------------------------------
__global__ __launch_bounds__(256)
void k_gates(const float* __restrict__ xt, const float* __restrict__ h,
             const float* __restrict__ ctx,
             const float* __restrict__ Wih, const float* __restrict__ Whh,
             const float* __restrict__ Wch, const float* __restrict__ bf,
             const float* __restrict__ biog, float* __restrict__ pre) {
  __shared__ __bf16 lds_a[32 * 40];
  __shared__ __bf16 lds_bt[64 * 40];
  const int tid  = threadIdx.x;
  const int lane = tid & 31, wid = tid >> 5;
  const int n0 = blockIdx.x * 64;
  const int wr = (wid & 1) * 16;
  const int wc = (wid >> 1) * 16;
  v8f acc = {0.f,0.f,0.f,0.f,0.f,0.f,0.f,0.f};

  for (int k0 = 0; k0 < 2048; k0 += 32) {
    const float* A; int ac, ald;
    const float* Bm; int br;
    if (k0 < 512)       { A = xt;  ac = k0;        ald = NEMB_;    Bm = Wih; br = k0; }
    else if (k0 < 1024) { A = h;   ac = k0 - 512;  ald = NHID_;    Bm = Whh; br = k0 - 512; }
    else                { A = ctx; ac = k0 - 1024; ald = SRC_HID_; Bm = Wch; br = k0 - 1024; }
#pragma unroll
    for (int j = 0; j < 2; ++j) {            // A: 32x32 (512 pairs)
      int p = tid + 256 * j;
      int r = p >> 4, c = (p & 15) << 1;
      const float2 v = *reinterpret_cast<const float2*>(&A[r * ald + ac + c]);
      st_bf16x2(&lds_a[r * 40 + c], v.x, v.y);
    }
#pragma unroll
    for (int j = 0; j < 4; ++j) {            // B: 32x64 transposed (1024 pairs)
      int p = tid + 256 * j;
      int n = p & 63, kr = (p >> 6) << 1;
      float b0 = Bm[(size_t)(br + kr)     * NG_ + n0 + n];
      float b1 = Bm[(size_t)(br + kr + 1) * NG_ + n0 + n];
      st_bf16x2(&lds_bt[n * 40 + kr], b0, b1);
    }
    __syncthreads();
    v16bf a = frag_a(&lds_a[wr * 40], 40, lane);
    v16bf b = frag_b(&lds_bt[wc * 40], 40, lane);
    acc = __builtin_amdgcn_wmma_f32_16x16x32_bf16(false, a, false, b, (short)0, acc, false, false);
    __syncthreads();
  }
  const int col = lane & 15;
  const int rb  = (lane >> 4) << 3;
#pragma unroll
  for (int r = 0; r < 8; ++r) {
    int b_ = wr + rb + r;
    int gn = n0 + wc + col;
    float bias = (gn < 512) ? bf[gn] : biog[gn - 512];
    pre[b_ * NG_ + gn] = acc[r] + bias;
  }
}

// ---------------------------------------------------------------------------
// LSTM gates + mask blending; h/c live in the output h_last/c_last slots.
// ---------------------------------------------------------------------------
__global__ __launch_bounds__(256)
void k_lstm(const float* __restrict__ pre, const float* __restrict__ mask, int t,
            float* __restrict__ h, float* __restrict__ c, float* __restrict__ hs) {
  const int idx = blockIdx.x * 256 + threadIdx.x;   // b*NHID + n
  const int b = idx >> 9, n = idx & 511;
  const float* pr = pre + b * NG_;
  const float f = pr[n], i = pr[512 + n], o = pr[1024 + n], g = pr[1536 + n];
  const float m = mask[b * T_STEPS + t];
  const float c_old = c[idx], h_old = h[idx];
  float c1 = sigmoidf_(f) * c_old + sigmoidf_(i) * fast_tanhf(g);
  c1 = c1 * m + c_old * (1.f - m);
  float h1 = sigmoidf_(o) * fast_tanhf(c1);
  h1 = h1 * m + h_old * (1.f - m);
  c[idx] = c1;
  h[idx] = h1;
  hs[(size_t)t * BATCH * NHID_ + idx] = h1 * m;
}

__global__ __launch_bounds__(256)
void k_init(const float* __restrict__ cc, float* __restrict__ h,
            float* __restrict__ c, float* __restrict__ cov) {
  const int i = blockIdx.x * 256 + threadIdx.x;
  h[i] = 0.f; c[i] = 0.f; cov[i] = cc[i];
}

// ---------------------------------------------------------------------------
extern "C" void kernel_launch(void* const* d_in, const int* in_sizes, int n_in,
                              void* d_out, int out_size, void* d_ws, size_t ws_size,
                              hipStream_t stream) {
  const float* x     = (const float*)d_in[0];
  const float* mask  = (const float*)d_in[1];
  const float* cc    = (const float*)d_in[2];
  const float* enc   = (const float*)d_in[3];
  const float* smask = (const float*)d_in[4];
  const float* Whw   = (const float*)d_in[5];
  const float* Whb   = (const float*)d_in[6];
  const float* Wsw   = (const float*)d_in[7];
  const float* Wcw   = (const float*)d_in[8];
  const float* vw    = (const float*)d_in[9];
  const float* Wih   = (const float*)d_in[10];
  const float* Whh   = (const float*)d_in[11];
  const float* Wch   = (const float*)d_in[12];
  const float* bf    = (const float*)d_in[13];
  const float* biog  = (const float*)d_in[14];

  // outputs (concatenated, return order)
  float* out    = (float*)d_out;
  float* hs     = out;                                     // (T,B,NHID)
  float* h      = hs + (size_t)T_STEPS * BATCH * NHID_;    // h_last: running h
  float* c      = h + BATCH * NHID_;                       // c_last: running c
  float* cov    = c + BATCH * NHID_;                       // cov_final: running
  float* attn   = cov + BATCH * S_LEN;                     // (B,T,S)
  float* logout = attn + (size_t)BATCH * T_STEPS * S_LEN;  // (B,T,S)

  // workspace
  float* ws    = (float*)d_ws;
  float* eh    = ws;                                       // 16384*512 f32
  float* es    = eh + (size_t)BATCH * S_LEN * ATTN_;       // 32*512
  float* logit = es + BATCH * ATTN_;                       // 32*512
  float* ctx   = logit + BATCH * S_LEN;                    // 32*1024
  float* pre   = ctx + BATCH * SRC_HID_;                   // 32*2048

  k_init<<<(BATCH * NHID_) / 256, 256, 0, stream>>>(cc, h, c, cov);
  k_gemm_eh<<<dim3(ATTN_ / 64, (BATCH * S_LEN) / 64), 256, 0, stream>>>(
      enc, Whw, Whb, smask, eh);

  for (int t = 0; t < T_STEPS; ++t) {
    k_es<<<ATTN_ / 64, 256, 0, stream>>>(h, Wsw, mask, t, es);
    k_attn<<<(BATCH * S_LEN) / 8, 256, 0, stream>>>(eh, es, cov, Wcw, vw, smask,
                                                    t, logit, logout);
    k_soft_ctx<<<BATCH, 512, 0, stream>>>(logit, smask, enc, t, cov, attn, ctx);
    k_gates<<<NG_ / 64, 256, 0, stream>>>(x + (size_t)t * BATCH * NEMB_, h, ctx,
                                          Wih, Whh, Wch, bf, biog, pre);
    k_lstm<<<(BATCH * NHID_) / 256, 256, 0, stream>>>(pre, mask, t, h, c, hs);
  }
}
